// SentimentClassificationMoE_53566832116404
// MI455X (gfx1250) — matrix operand order repaired and checked by reading, not compiled, not run on hardware
//
#include <hip/hip_runtime.h>
#include <hip/hip_bf16.h>
#include <math.h>

#define VOCAB 100000
#define DIM   512
#define FF    2048
#define NE    8
#define OUTD  2
#define BATCH 4096
#define SEQ   200
#define TPAD  (BATCH + NE * 16)   /* 4224: per-expert segments padded to 16 */
#define MT_MAX (TPAD / 16)        /* 264 row tiles */
#define SUBN  4                   /* 16x16 subtiles per wave -> wave tile 16x64 */

typedef __attribute__((ext_vector_type(16))) __bf16 v16bf;
typedef __attribute__((ext_vector_type(8)))  float  v8f;

union AB16 { v16bf v; uint4 q[2]; unsigned short u[16]; };

#if __has_builtin(__builtin_amdgcn_sched_barrier)
#define SCHED_FENCE() __builtin_amdgcn_sched_barrier(0)
#else
#define SCHED_FENCE() __threadfence_block()  /* weaker fallback */
#endif

__device__ __forceinline__ unsigned short f2bf(float f) {
    unsigned int u = __float_as_uint(f);
    unsigned int r = u + 0x7FFFu + ((u >> 16) & 1u);  // round-to-nearest-even
    return (unsigned short)(r >> 16);
}

// ---------------- 1. embedding gather + mean pool: x[B,S] -> h[B,512] (f32 + bf16)
__global__ void k_embed_pool(const int* __restrict__ x, const float* __restrict__ emb,
                             float* __restrict__ h, unsigned short* __restrict__ hb) {
    int b = blockIdx.x;
    int t = threadIdx.x;                       // 128 threads, one float4 column each
    const int* xb = x + (size_t)b * SEQ;
    float4 acc = make_float4(0.f, 0.f, 0.f, 0.f);
    for (int s = 0; s < SEQ; ++s) {
        if (s + 4 < SEQ) {                     // keep gather rows in flight
            int ptok = xb[s + 4];
            __builtin_prefetch((const float4*)(emb + (size_t)ptok * DIM) + t, 0, 0);
        }
        int tok = xb[s];                       // uniform scalar load
        float4 v = ((const float4*)(emb + (size_t)tok * DIM))[t];
        acc.x += v.x; acc.y += v.y; acc.z += v.z; acc.w += v.w;
    }
    const float inv = 1.0f / (float)SEQ;
    acc.x *= inv; acc.y *= inv; acc.z *= inv; acc.w *= inv;
    ((float4*)(h + (size_t)b * DIM))[t] = acc;
    unsigned int lo = (unsigned int)f2bf(acc.x) | ((unsigned int)f2bf(acc.y) << 16);
    unsigned int hi = (unsigned int)f2bf(acc.z) | ((unsigned int)f2bf(acc.w) << 16);
    ((uint2*)(hb + (size_t)b * DIM))[t] = make_uint2(lo, hi);
}

// ---------------- 2. init scratch (perm = -1, counts = 0)
__global__ void k_init(int* __restrict__ perm, int* __restrict__ counts) {
    int i = blockIdx.x * blockDim.x + threadIdx.x;
    if (i < TPAD) perm[i] = -1;
    if (i < NE)   counts[i] = 0;
}

// ---------------- 3. top-1 gating: softmax(h@Wg) -> idx, gate; histogram counts
__global__ void k_gate(const float* __restrict__ h, const float* __restrict__ Wg,
                       int* __restrict__ idx, float* __restrict__ gate,
                       int* __restrict__ counts) {
    int wave = threadIdx.x >> 5, lane = threadIdx.x & 31;
    int b = blockIdx.x * (blockDim.x >> 5) + wave;
    if (b >= BATCH) return;
    float acc[NE];
    #pragma unroll
    for (int e = 0; e < NE; ++e) acc[e] = 0.f;
    const float* hp = h + (size_t)b * DIM;
    for (int d = lane; d < DIM; d += 32) {
        float hv = hp[d];
        float4 w0 = ((const float4*)(Wg + (size_t)d * NE))[0];
        float4 w1 = ((const float4*)(Wg + (size_t)d * NE))[1];
        acc[0] += hv * w0.x; acc[1] += hv * w0.y; acc[2] += hv * w0.z; acc[3] += hv * w0.w;
        acc[4] += hv * w1.x; acc[5] += hv * w1.y; acc[6] += hv * w1.z; acc[7] += hv * w1.w;
    }
    #pragma unroll
    for (int off = 16; off >= 1; off >>= 1)
        #pragma unroll
        for (int e = 0; e < NE; ++e) acc[e] += __shfl_down(acc[e], off, 32);
    if (lane == 0) {
        float m = acc[0]; int am = 0;
        #pragma unroll
        for (int e = 1; e < NE; ++e) if (acc[e] > m) { m = acc[e]; am = e; }
        float s = 0.f;
        #pragma unroll
        for (int e = 0; e < NE; ++e) s += __expf(acc[e] - m);
        idx[b] = am;
        gate[b] = 1.0f / s;           // exp(max-max)/sum
        atomicAdd(&counts[am], 1);
    }
}

// ---------------- 4. padded prefix-sum over expert counts
__global__ void k_offsets(const int* __restrict__ counts, int* __restrict__ off_pad,
                          int* __restrict__ cursor) {
    int off = 0;
    for (int e = 0; e < NE; ++e) {
        off_pad[e] = off; cursor[e] = off;
        off += (counts[e] + 15) & ~15;
    }
    off_pad[NE] = off;
}

// ---------------- 5. scatter tokens into expert-grouped permutation
__global__ void k_scatter(const int* __restrict__ idx, int* __restrict__ cursor,
                          int* __restrict__ perm) {
    int b = blockIdx.x * blockDim.x + threadIdx.x;
    if (b >= BATCH) return;
    int slot = atomicAdd(&cursor[idx[b]], 1);
    perm[slot] = b;
}

// ---------------- 6. f32 [E][R][C] -> bf16 transposed [E][C][R] (for WMMA B fragments)
__global__ void k_tconv(const float* __restrict__ src, unsigned short* __restrict__ dst,
                        int R, int C) {
    __shared__ unsigned short tile[32][33];
    int e = blockIdx.z;
    int c0 = blockIdx.x * 32, r0 = blockIdx.y * 32;
    const float* s = src + (size_t)e * R * C;
    unsigned short* d = dst + (size_t)e * R * C;
    int tx = threadIdx.x, ty = threadIdx.y;    // (32,8)
    #pragma unroll
    for (int j = 0; j < 32; j += 8)
        tile[ty + j][tx] = f2bf(s[(size_t)(r0 + ty + j) * C + c0 + tx]);
    __syncthreads();
    #pragma unroll
    for (int j = 0; j < 32; j += 8)
        d[(size_t)(c0 + ty + j) * R + r0 + tx] = tile[tx][ty + j];
}

// fragment loaders -------------------------------------------------------------
__device__ __forceinline__ void load_a_frag(AB16& a, const unsigned short* As_row,
                                            int k0, int lhalf) {
    // A 16x32 bf16 layout: lane-lo K{0..7,16..23}, lane-hi K{8..15,24..31}
    a.q[0] = *(const uint4*)(As_row + k0 + lhalf * 8);
    a.q[1] = *(const uint4*)(As_row + k0 + lhalf * 8 + 16);
}
__device__ __forceinline__ void load_b_frag(AB16& b, const unsigned short* brow,
                                            int k0, int lhalf) {
    // B 32x16 bf16 layout: lane-lo K{0..7,16..23}, lane-hi K{8..15,24..31}
    const unsigned short* bp = brow + k0 + lhalf * 8;
    b.q[0] = *(const uint4*)(bp);
    b.q[1] = *(const uint4*)(bp + 16);
}

// pipelined K=512 panel: loads for step k+1 are pinned above WMMAs for step k
__device__ __forceinline__ void gemm_panel_512(v8f c[SUBN], const unsigned short* As_row,
                                               const unsigned short* const brow[SUBN],
                                               int lhalf) {
    AB16 a_cur, b_cur[SUBN];
    load_a_frag(a_cur, As_row, 0, lhalf);
    #pragma unroll
    for (int s = 0; s < SUBN; ++s) load_b_frag(b_cur[s], brow[s], 0, lhalf);
    #pragma unroll
    for (int k0 = 0; k0 < 512 - 32; k0 += 32) {
        AB16 a_nxt, b_nxt[SUBN];
        load_a_frag(a_nxt, As_row, k0 + 32, lhalf);
        #pragma unroll
        for (int s = 0; s < SUBN; ++s) load_b_frag(b_nxt[s], brow[s], k0 + 32, lhalf);
        SCHED_FENCE();   // keep next-step loads issued above this step's WMMAs
        #pragma unroll
        for (int s = 0; s < SUBN; ++s)
            c[s] = __builtin_amdgcn_wmma_f32_16x16x32_bf16(false, a_cur.v, false, b_cur[s].v,
                                                           (short)0, c[s], false, false);
        SCHED_FENCE();
        a_cur = a_nxt;
        #pragma unroll
        for (int s = 0; s < SUBN; ++s) b_cur[s] = b_nxt[s];
    }
    #pragma unroll
    for (int s = 0; s < SUBN; ++s)
        c[s] = __builtin_amdgcn_wmma_f32_16x16x32_bf16(false, a_cur.v, false, b_cur[s].v,
                                                       (short)0, c[s], false, false);
}

// ---------------- 7. grouped GEMM pass 1: hid = relu(Hperm @ W1[e] + b1[e]) (bf16 out)
// block = 128 threads (4 waves); block tile 16(M) x 256(N); wave tile 16x64
__global__ void k_ffn1(const unsigned short* __restrict__ hb,
                       const unsigned short* __restrict__ w1t,   // [E][FF][DIM]
                       const float* __restrict__ b1,
                       const int* __restrict__ perm,
                       const int* __restrict__ off_pad,
                       unsigned short* __restrict__ hidp) {      // [TPAD][FF]
    __shared__ unsigned short As[16][DIM + 8];
    __shared__ int toks[16];
    __shared__ int se_s;
    int tid = threadIdx.x;
    int r0 = blockIdx.x * 16;
    if (tid < 16) toks[tid] = perm[r0 + tid];
    if (tid == 0) {
        int se = NE - 1;
        for (int e = 0; e < NE; ++e) if (r0 < off_pad[e + 1]) { se = e; break; }
        se_s = se;
    }
    __syncthreads();
    for (int i = tid; i < 16 * 64; i += 128) {           // 16 rows x 64 chunks of 8 halves
        int row = i >> 6, c8 = i & 63;
        int tok = toks[row];
        uint4 v = make_uint4(0u, 0u, 0u, 0u);
        if (tok >= 0) v = *(const uint4*)(hb + (size_t)tok * DIM + c8 * 8);
        *(uint4*)(&As[row][c8 * 8]) = v;
    }
    __syncthreads();
    int se = se_s;
    int wave = tid >> 5, lane = tid & 31;
    int n0 = blockIdx.y * (SUBN * 64) + wave * (SUBN * 16);
    int lhalf = lane >> 4, lm = lane & 15;
    const unsigned short* As_row = &As[lm][0];
    const unsigned short* brow[SUBN];
    #pragma unroll
    for (int s = 0; s < SUBN; ++s)
        brow[s] = w1t + ((size_t)se * FF + (n0 + s * 16 + lm)) * DIM;
    v8f c[SUBN];
    #pragma unroll
    for (int s = 0; s < SUBN; ++s) c[s] = (v8f){};
    gemm_panel_512(c, As_row, brow, lhalf);
    #pragma unroll
    for (int s = 0; s < SUBN; ++s) {
        int n = n0 + s * 16 + lm;
        float bias = b1[(size_t)se * FF + n];
        #pragma unroll
        for (int v = 0; v < 8; ++v) {
            int m = v + 8 * lhalf;             // C/D layout: VGPR v -> row v (+8 for hi lanes)
            float val = c[s][v] + bias;
            val = val > 0.f ? val : 0.f;
            hidp[(size_t)(r0 + m) * FF + n] = f2bf(val);
        }
    }
}

// ---------------- 8. grouped GEMM pass 2: moe[tok] = (hid @ W2[e] + b2[e]) * gate[tok]
// block tile 16(M) x 256(N); K = 2048 chunked 4x512 through LDS
__global__ void k_ffn2(const unsigned short* __restrict__ hidp,
                       const unsigned short* __restrict__ w2t,   // [E][DIM][FF]
                       const float* __restrict__ b2,
                       const float* __restrict__ gate,
                       const int* __restrict__ perm,
                       const int* __restrict__ off_pad,
                       float* __restrict__ moe) {
    __shared__ unsigned short As[16][512 + 8];
    __shared__ int toks[16];
    __shared__ float gts[16];
    __shared__ int se_s;
    int tid = threadIdx.x;
    int r0 = blockIdx.x * 16;
    if (tid < 16) {
        int tok = perm[r0 + tid];
        toks[tid] = tok;
        gts[tid] = (tok >= 0) ? gate[tok] : 0.f;
    }
    if (tid == 0) {
        int se = NE - 1;
        for (int e = 0; e < NE; ++e) if (r0 < off_pad[e + 1]) { se = e; break; }
        se_s = se;
    }
    __syncthreads();
    int se = se_s;
    int wave = tid >> 5, lane = tid & 31;
    int n0 = blockIdx.y * (SUBN * 64) + wave * (SUBN * 16);
    int lhalf = lane >> 4, lm = lane & 15;
    const unsigned short* As_row = &As[lm][0];
    const unsigned short* browk[SUBN];
    #pragma unroll
    for (int s = 0; s < SUBN; ++s)
        browk[s] = w2t + ((size_t)se * DIM + (n0 + s * 16 + lm)) * FF;
    v8f c[SUBN];
    #pragma unroll
    for (int s = 0; s < SUBN; ++s) c[s] = (v8f){};
    for (int kc = 0; kc < FF; kc += 512) {               // K chunked through LDS
        __syncthreads();
        for (int i = tid; i < 16 * 64; i += 128) {
            int row = i >> 6, c8 = i & 63;
            *(uint4*)(&As[row][c8 * 8]) =
                *(const uint4*)(hidp + (size_t)(r0 + row) * FF + kc + c8 * 8);
        }
        __syncthreads();
        const unsigned short* brow[SUBN];
        #pragma unroll
        for (int s = 0; s < SUBN; ++s) brow[s] = browk[s] + kc;
        gemm_panel_512(c, As_row, brow, lhalf);
    }
    #pragma unroll
    for (int s = 0; s < SUBN; ++s) {
        int n = n0 + s * 16 + lm;
        float bias = b2[(size_t)se * DIM + n];
        #pragma unroll
        for (int v = 0; v < 8; ++v) {
            int m = v + 8 * lhalf;
            int tok = toks[m];
            if (tok >= 0) moe[(size_t)tok * DIM + n] = (c[s][v] + bias) * gts[m];
        }
    }
}

// ---------------- 9. final classifier + log_softmax over 2 classes
__global__ void k_fc_lsm(const float* __restrict__ moe, const float* __restrict__ Wfc,
                         const float* __restrict__ bfc, float* __restrict__ out) {
    int wave = threadIdx.x >> 5, lane = threadIdx.x & 31;
    int b = blockIdx.x * (blockDim.x >> 5) + wave;
    if (b >= BATCH) return;
    float a0 = 0.f, a1 = 0.f;
    const float* mb = moe + (size_t)b * DIM;
    for (int d = lane; d < DIM; d += 32) {
        float m = mb[d];
        float2 w = *(const float2*)(Wfc + d * 2);
        a0 += m * w.x; a1 += m * w.y;
    }
    #pragma unroll
    for (int off = 16; off >= 1; off >>= 1) {
        a0 += __shfl_down(a0, off, 32);
        a1 += __shfl_down(a1, off, 32);
    }
    if (lane == 0) {
        float z0 = a0 + bfc[0], z1 = a1 + bfc[1];
        float mx = fmaxf(z0, z1);
        float lse = mx + __logf(__expf(z0 - mx) + __expf(z1 - mx));
        out[(size_t)b * 2]     = z0 - lse;
        out[(size_t)b * 2 + 1] = z1 - lse;
    }
}

extern "C" void kernel_launch(void* const* d_in, const int* in_sizes, int n_in,
                              void* d_out, int out_size, void* d_ws, size_t ws_size,
                              hipStream_t stream) {
    const int*   x    = (const int*)d_in[0];
    const float* emb  = (const float*)d_in[1];
    const float* Wg   = (const float*)d_in[2];
    const float* W1   = (const float*)d_in[3];
    const float* b1   = (const float*)d_in[4];
    const float* W2   = (const float*)d_in[5];
    const float* b2   = (const float*)d_in[6];
    const float* Wfc  = (const float*)d_in[7];
    const float* bfc  = (const float*)d_in[8];
    float* out = (float*)d_out;

    char* p = (char*)d_ws;
    auto alloc = [&](size_t bytes) -> char* {
        char* r = p;
        p += (bytes + 255) & ~(size_t)255;
        return r;
    };
    float*          h      = (float*)alloc((size_t)BATCH * DIM * 4);
    unsigned short* hbf    = (unsigned short*)alloc((size_t)BATCH * DIM * 2);
    unsigned short* w1t    = (unsigned short*)alloc((size_t)NE * FF * DIM * 2);
    unsigned short* w2t    = (unsigned short*)alloc((size_t)NE * FF * DIM * 2);
    unsigned short* hidp   = (unsigned short*)alloc((size_t)TPAD * FF * 2);
    float*          moe    = (float*)alloc((size_t)BATCH * DIM * 4);
    int*            idx    = (int*)alloc((size_t)BATCH * 4);
    float*          gate   = (float*)alloc((size_t)BATCH * 4);
    int*            perm   = (int*)alloc((size_t)TPAD * 4);
    int*            counts = (int*)alloc(64);
    int*            offp   = (int*)alloc(64);
    int*            cursor = (int*)alloc(64);

    k_init<<<(TPAD + 255) / 256, 256, 0, stream>>>(perm, counts);
    k_embed_pool<<<BATCH, 128, 0, stream>>>(x, emb, h, hbf);
    k_tconv<<<dim3(FF / 32, DIM / 32, NE), dim3(32, 8), 0, stream>>>(W1, w1t, DIM, FF);
    k_tconv<<<dim3(DIM / 32, FF / 32, NE), dim3(32, 8), 0, stream>>>(W2, w2t, FF, DIM);
    k_gate<<<BATCH / 8, 256, 0, stream>>>(h, Wg, idx, gate, counts);
    k_offsets<<<1, 1, 0, stream>>>(counts, offp, cursor);
    k_scatter<<<BATCH / 256, 256, 0, stream>>>(idx, cursor, perm);
    k_ffn1<<<dim3(MT_MAX, FF / (SUBN * 64)), 128, 0, stream>>>(hbf, w1t, b1, perm, offp, hidp);
    k_ffn2<<<dim3(MT_MAX, DIM / (SUBN * 64)), 128, 0, stream>>>(hidp, w2t, b2, gate, perm, offp, moe);
    k_fc_lsm<<<BATCH / 8, 256, 0, stream>>>(moe, Wfc, bfc, out);
}
